// MultiHeadAttentionLayer_4973572129389
// MI455X (gfx1250) — matrix operand order repaired and checked by reading, not compiled, not run on hardware
//
#include <hip/hip_runtime.h>

// ---------------------------------------------------------------------------
// MHA for MI455X (gfx1250): bf16 WMMA, flash-attention softmax, bit-packed
// mask, async-to-LDS double-buffered K/V staging.
// ---------------------------------------------------------------------------

typedef __attribute__((ext_vector_type(16))) __bf16 v16bf;
typedef __attribute__((ext_vector_type(8)))  float  v8f;
typedef __attribute__((ext_vector_type(4)))  int    v4i;

union Frag {
  v16bf v;
  uint4 q[2];
  unsigned short u[16];
};

static constexpr int B  = 2;
static constexpr int S  = 2048;
static constexpr int E  = 1024;
static constexpr int H  = 16;
static constexpr int D  = 64;
static constexpr int HD = H * D;        // 1024
static constexpr int SW = S / 32;       // mask words per row = 64
static constexpr float NEGV = -1e9f;

__device__ __forceinline__ unsigned short f2bf(float f) {
  unsigned int u = __float_as_uint(f);
  u += 0x7fffu + ((u >> 16) & 1u);      // round-to-nearest-even
  return (unsigned short)(u >> 16);
}

// ---- gfx1250 async global->LDS copy (16B per lane) ------------------------
// Probe-confirmed: __builtin_amdgcn_global_load_async_to_lds_b128 expects
// (addrspace(1) v4i* src, addrspace(3) v4i* dst, int offset, int cpol).
typedef __attribute__((address_space(1))) v4i* gv4i_p;
typedef __attribute__((address_space(3))) v4i* lv4i_p;

__device__ __forceinline__ void async_cp16(unsigned short* l, const unsigned short* g) {
#if defined(__has_builtin) && __has_builtin(__builtin_amdgcn_global_load_async_to_lds_b128)
  __builtin_amdgcn_global_load_async_to_lds_b128(
      (gv4i_p)(void*)const_cast<unsigned short*>(g),
      (lv4i_p)(void*)l, 0, 0);
#else
  unsigned loff = (unsigned)(uintptr_t)(lv4i_p)(void*)l;   // LDS byte offset
  unsigned long long ga = (unsigned long long)(uintptr_t)g;
  asm volatile("global_load_async_to_lds_b128 %0, %1, off"
               :: "v"(loff), "v"(ga) : "memory");
#endif
}

__device__ __forceinline__ void wait_async0() {
#if defined(__has_builtin) && __has_builtin(__builtin_amdgcn_s_wait_asynccnt)
  __builtin_amdgcn_s_wait_asynccnt(0);
#else
  asm volatile("s_wait_asynccnt 0x0" ::: "memory");
#endif
}

// ---------------------------------------------------------------------------
// Kernel 1: convert x to bf16; convert+transpose weights; bit-pack the mask
// (one uint32 per 32 keys via wave32 ballot).
//   Wq/Wk/Wv [H,E,D] -> [H,D,E] bf16 ;  Wo [HD,E] -> [E,HD] bf16
// ---------------------------------------------------------------------------
__global__ __launch_bounds__(256) void cvt_kernel(
    const float* __restrict__ x, const int* __restrict__ mask,
    const float* __restrict__ Wq, const float* __restrict__ Wk,
    const float* __restrict__ Wv, const float* __restrict__ Wo,
    unsigned short* __restrict__ xb,
    unsigned short* __restrict__ wtq, unsigned short* __restrict__ wtk,
    unsigned short* __restrict__ wtv, unsigned short* __restrict__ wot,
    unsigned int* __restrict__ mb) {
  long i = (long)blockIdx.x * blockDim.x + threadIdx.x;
  const long NX = (long)B * S * E;   // 4M
  const long NW = (long)H * E * D;   // 1M
  const long NO = (long)HD * E;      // 1M
  const long NM = (long)S * S;       // 4M
  if (i < NX) { xb[i] = f2bf(x[i]); return; }
  i -= NX;
  if (i < NW) { long h = i/(E*D), r = i%(E*D), e = r/D, d = r%D;
                wtq[(h*D + d)*E + e] = f2bf(Wq[i]); return; }
  i -= NW;
  if (i < NW) { long h = i/(E*D), r = i%(E*D), e = r/D, d = r%D;
                wtk[(h*D + d)*E + e] = f2bf(Wk[i]); return; }
  i -= NW;
  if (i < NW) { long h = i/(E*D), r = i%(E*D), e = r/D, d = r%D;
                wtv[(h*D + d)*E + e] = f2bf(Wv[i]); return; }
  i -= NW;
  if (i < NO) { long hd = i/E, e = i%E;
                wot[e*HD + hd] = f2bf(Wo[i]); return; }
  i -= NO;
  if (i < NM) {                       // wave32 ballot: bit j <-> key (base+j)
    unsigned word = (unsigned)__ballot(mask[i] != 0);
    if ((i & 31) == 0) mb[i >> 5] = word;
    return;
  }
}

// ---------------------------------------------------------------------------
// Kernel 2: Q/K/V projections. One wave per 16x16 tile of one of {Q,K,V}.
// Q,K row-major [B,H,S,D] bf16 (Q pre-scaled by 1/sqrt(D));
// V transposed [B,H,D,S] bf16.
// ---------------------------------------------------------------------------
__global__ __launch_bounds__(256) void qkv_kernel(
    const unsigned short* __restrict__ xb,
    const unsigned short* __restrict__ wtq, const unsigned short* __restrict__ wtk,
    const unsigned short* __restrict__ wtv,
    const float* __restrict__ bq, const float* __restrict__ bk,
    const float* __restrict__ bv,
    unsigned short* __restrict__ Qb, unsigned short* __restrict__ Kb,
    unsigned short* __restrict__ Vt) {
  const int lane = threadIdx.x & 31;
  const int w = blockIdx.x * (blockDim.x >> 5) + (threadIdx.x >> 5);
  const int which = w >> 14;           // 0=Q 1=K 2=V (16384 tiles per matrix)
  const int t  = w & 16383;
  const int dt = t & 3;                // D/16
  const int st = (t >> 2) & 127;       // S/16
  const int h  = (t >> 9) & 15;
  const int b  = t >> 13;

  const unsigned short* wt   = (which == 0) ? wtq : (which == 1) ? wtk : wtv;
  const float*          bias = (which == 0) ? bq  : (which == 1) ? bk  : bv;

  const int n = lane & 15, half = lane >> 4;
  const int srow = st * 16 + n;        // A-frag row (M)
  const int dcol = dt * 16 + n;        // B-frag / C column (N)
  const unsigned short* aRow = xb + ((long)b * S + srow) * E;
  const unsigned short* bRow = wt + ((long)h * D + dcol) * E;

  v8f acc = {};
  for (int e0 = 0; e0 < E; e0 += 32) {
    Frag a, bf;
    a.q[0]  = *(const uint4*)(aRow + e0 + half * 8);
    a.q[1]  = *(const uint4*)(aRow + e0 + half * 8 + 16);
    bf.q[0] = *(const uint4*)(bRow + e0 + half * 8);
    bf.q[1] = *(const uint4*)(bRow + e0 + half * 8 + 16);
    acc = __builtin_amdgcn_wmma_f32_16x16x32_bf16(
        false, a.v, false, bf.v, (short)0, acc, false, false);
  }
  const float bvv   = bias[h * D + dcol];
  const float scale = (which == 0) ? 0.125f : 1.0f;   // fold 1/sqrt(D) into Q
#pragma unroll
  for (int r = 0; r < 8; ++r) acc[r] = (acc[r] + bvv) * scale;

  const long bh = (long)b * H + h;
  if (which == 2) {                     // V transposed: [B,H,D,S]
    const int sstart = st * 16 + half * 8;
    unsigned int p[4];
#pragma unroll
    for (int r = 0; r < 4; ++r)
      p[r] = (unsigned)f2bf(acc[2 * r]) | ((unsigned)f2bf(acc[2 * r + 1]) << 16);
    uint4 pk; pk.x = p[0]; pk.y = p[1]; pk.z = p[2]; pk.w = p[3];
    *(uint4*)(Vt + (bh * D + dcol) * S + sstart) = pk;
  } else {                              // Q,K row-major: [B,H,S,D]
    unsigned short* dst = (which == 0) ? Qb : Kb;
#pragma unroll
    for (int r = 0; r < 8; ++r) {
      const int s = st * 16 + half * 8 + r;
      dst[(bh * S + s) * D + dt * 16 + n] = f2bf(acc[r]);
    }
  }
}

// ---------------------------------------------------------------------------
// Kernel 3: flash attention. Block = 4 waves = 4 q-tiles of the same (b,h).
// K/V streamed in 32-key chunks via double-buffered async-to-LDS staging;
// WMMA fragments read from LDS; bit-packed mask; online softmax with
// 16-lane shuffle reductions; P transposed C->A layout via wave-private LDS.
// ---------------------------------------------------------------------------
__global__ __launch_bounds__(128) void attn_kernel(
    const unsigned short* __restrict__ Qb, const unsigned short* __restrict__ Kb,
    const unsigned short* __restrict__ Vt, const unsigned int* __restrict__ mb,
    unsigned short* __restrict__ ctxb) {
  __shared__ unsigned short kst[2][32 * 64];   // [k-local][d]   4KB x2
  __shared__ unsigned short vst[2][64 * 32];   // [d][k-local]   4KB x2
  __shared__ unsigned short pst[4][16 * 32];   // per-wave P transpose

  const int tid  = threadIdx.x;
  const int lane = tid & 31;
  const int widx = tid >> 5;
  const int qg = blockIdx.x & 31;              // 32 q-groups per head
  const int h  = (blockIdx.x >> 5) & 15;
  const int b  = blockIdx.x >> 9;
  const int qt = qg * 4 + widx;
  const long bh = (long)b * H + h;
  const int n = lane & 15, half = lane >> 4;

  const unsigned short* Kbh = Kb + bh * S * D;
  const unsigned short* Vbh = Vt + bh * D * S;

  // Q A-fragments (pre-scaled by 1/8), kept in registers for the whole loop
  Frag qf[2];
  {
    const unsigned short* qp = Qb + (bh * S + qt * 16 + n) * D;
#pragma unroll
    for (int dc = 0; dc < 2; ++dc) {
      qf[dc].q[0] = *(const uint4*)(qp + dc * 32 + half * 8);
      qf[dc].q[1] = *(const uint4*)(qp + dc * 32 + half * 8 + 16);
    }
  }

  float m[8], l[8];
#pragma unroll
  for (int r = 0; r < 8; ++r) { m[r] = -3.0e38f; l[r] = 0.0f; }
  v8f cacc[4] = {};
  unsigned short* pl = pst[widx];

  // cooperative stage of one 32-key chunk: K = flat 4KB, V = 64 strided rows
  auto stage = [&](int c) {
    const unsigned short* kg = Kbh + (long)c * 32 * D;   // contiguous 4KB
    const unsigned short* vg = Vbh + c * 32;             // 64 rows x 64B
    unsigned short* lk = kst[c & 1];
    unsigned short* lv = vst[c & 1];
#pragma unroll
    for (int i = 0; i < 2; ++i) {
      const int t = tid + i * 128;                       // 0..255
      async_cp16(lk + t * 8, kg + t * 8);
      const int d = t >> 2, p = t & 3;
      async_cp16(lv + d * 32 + p * 8, vg + (long)d * S + p * 8);
    }
  };

  stage(0);
  for (int c = 0; c < S / 32; ++c) {
    wait_async0();
    __syncthreads();                    // chunk c staged, all waves past c-1
    if (c + 1 < S / 32) stage(c + 1);   // overlap next copy with compute
    const int k0 = c * 32;
    const unsigned short* lk = kst[c & 1];
    const unsigned short* lv = vst[c & 1];

    // ---- bit-packed mask words (uniform per row-half) ----
    unsigned wrd[8];
#pragma unroll
    for (int r = 0; r < 8; ++r)
      wrd[r] = mb[(long)(qt * 16 + half * 8 + r) * SW + (k0 >> 5)];

    // ---- scores: two 16x16 tiles, 2 k-steps each, K-frags from LDS ----
    v8f sc[2];
#pragma unroll
    for (int nt = 0; nt < 2; ++nt) {
      v8f a = {};
#pragma unroll
      for (int dc = 0; dc < 2; ++dc) {
        Frag kf;
        const unsigned short* kp = lk + (nt * 16 + n) * 64 + dc * 32 + half * 8;
        kf.q[0] = *(const uint4*)(kp);
        kf.q[1] = *(const uint4*)(kp + 16);
        a = __builtin_amdgcn_wmma_f32_16x16x32_bf16(
            false, qf[dc].v, false, kf.v, (short)0, a, false, false);
      }
      sc[nt] = a;
    }
    // ---- mask (scale already folded into Q) ----
#pragma unroll
    for (int nt = 0; nt < 2; ++nt)
#pragma unroll
      for (int r = 0; r < 8; ++r)
        sc[nt][r] = ((wrd[r] >> (nt * 16 + n)) & 1u) ? sc[nt][r] : NEGV;

    // ---- online softmax: row stats across the 16 lanes of a half ----
    float p0[8], p1[8];
#pragma unroll
    for (int r = 0; r < 8; ++r) {
      float cm = fmaxf(sc[0][r], sc[1][r]);
      cm = fmaxf(cm, __shfl_xor(cm, 1, 16));
      cm = fmaxf(cm, __shfl_xor(cm, 2, 16));
      cm = fmaxf(cm, __shfl_xor(cm, 4, 16));
      cm = fmaxf(cm, __shfl_xor(cm, 8, 16));
      const float mn   = fmaxf(m[r], cm);
      const float corr = __expf(m[r] - mn);
      p0[r] = __expf(sc[0][r] - mn);
      p1[r] = __expf(sc[1][r] - mn);
      float rs = p0[r] + p1[r];
      rs += __shfl_xor(rs, 1, 16);
      rs += __shfl_xor(rs, 2, 16);
      rs += __shfl_xor(rs, 4, 16);
      rs += __shfl_xor(rs, 8, 16);
      l[r] = l[r] * corr + rs;
      m[r] = mn;
#pragma unroll
      for (int d2 = 0; d2 < 4; ++d2) cacc[d2][r] *= corr;
    }
    // ---- transpose P (C layout -> A layout) via wave-private LDS ----
#pragma unroll
    for (int r = 0; r < 8; ++r) {
      const int row = half * 8 + r;
      pl[row * 32 + n]      = f2bf(p0[r]);
      pl[row * 32 + 16 + n] = f2bf(p1[r]);
    }
    Frag pf;
    pf.q[0] = *(const uint4*)(pl + n * 32 + half * 8);
    pf.q[1] = *(const uint4*)(pl + n * 32 + half * 8 + 16);
    // ---- ctx += P @ V, V-frags from LDS ----
#pragma unroll
    for (int d2 = 0; d2 < 4; ++d2) {
      Frag vf;
      const unsigned short* vp = lv + (d2 * 16 + n) * 32 + half * 8;
      vf.q[0] = *(const uint4*)(vp);
      vf.q[1] = *(const uint4*)(vp + 16);
      cacc[d2] = __builtin_amdgcn_wmma_f32_16x16x32_bf16(
          false, pf.v, false, vf.v, (short)0, cacc[d2], false, false);
    }
  }

  // ---- normalize and store ctx as bf16 [B, S, H*D] ----
#pragma unroll
  for (int r = 0; r < 8; ++r) {
    const float inv = (l[r] > 0.0f) ? (1.0f / l[r]) : 0.0f;
    const int s = qt * 16 + half * 8 + r;
#pragma unroll
    for (int d2 = 0; d2 < 4; ++d2) {
      ctxb[((long)b * S + s) * HD + h * D + d2 * 16 + n] =
          f2bf(cacc[d2][r] * inv);
    }
  }
}

// ---------------------------------------------------------------------------
// Kernel 4: output projection  out[B*S, E] = ctx_bf16 @ Wo + bo   (fp32 out)
// ---------------------------------------------------------------------------
__global__ __launch_bounds__(256) void oproj_kernel(
    const unsigned short* __restrict__ ctxb, const unsigned short* __restrict__ wot,
    const float* __restrict__ bo, float* __restrict__ out) {
  const int lane = threadIdx.x & 31;
  const int w  = blockIdx.x * (blockDim.x >> 5) + (threadIdx.x >> 5);
  const int et = w & 63;               // E/16
  const int st = w >> 6;               // (B*S)/16 = 256 strips
  const int n = lane & 15, half = lane >> 4;

  const unsigned short* aRow = ctxb + ((long)st * 16 + n) * HD;
  const unsigned short* bRow = wot + ((long)et * 16 + n) * HD;

  v8f acc = {};
  for (int k0 = 0; k0 < HD; k0 += 32) {
    Frag a, bf;
    a.q[0]  = *(const uint4*)(aRow + k0 + half * 8);
    a.q[1]  = *(const uint4*)(aRow + k0 + half * 8 + 16);
    bf.q[0] = *(const uint4*)(bRow + k0 + half * 8);
    bf.q[1] = *(const uint4*)(bRow + k0 + half * 8 + 16);
    acc = __builtin_amdgcn_wmma_f32_16x16x32_bf16(
        false, a.v, false, bf.v, (short)0, acc, false, false);
  }
  const float bias = bo[et * 16 + n];
#pragma unroll
  for (int r = 0; r < 8; ++r) {
    const int row = st * 16 + half * 8 + r;
    out[(long)row * E + et * 16 + n] = acc[r] + bias;
  }
}

// ---------------------------------------------------------------------------
extern "C" void kernel_launch(void* const* d_in, const int* in_sizes, int n_in,
                              void* d_out, int out_size, void* d_ws, size_t ws_size,
                              hipStream_t stream) {
  (void)in_sizes; (void)n_in; (void)out_size; (void)ws_size;
  const float* x    = (const float*)d_in[0];
  const int*   mask = (const int*)  d_in[1];
  const float* Wq   = (const float*)d_in[2];
  const float* bq   = (const float*)d_in[3];
  const float* Wk   = (const float*)d_in[4];
  const float* bk   = (const float*)d_in[5];
  const float* Wv   = (const float*)d_in[6];
  const float* bv   = (const float*)d_in[7];
  const float* Wo   = (const float*)d_in[8];
  const float* bo   = (const float*)d_in[9];
  float* out = (float*)d_out;

  char* ws = (char*)d_ws;
  size_t off = 0;
  auto alloc = [&](size_t bytes) -> void* {
    void* p = ws + off;
    off += (bytes + 255) & ~(size_t)255;
    return p;
  };
  unsigned short* xb  = (unsigned short*)alloc((size_t)B * S * E * 2);     // 8 MB
  unsigned short* wtq = (unsigned short*)alloc((size_t)H * E * D * 2);     // 2 MB
  unsigned short* wtk = (unsigned short*)alloc((size_t)H * E * D * 2);
  unsigned short* wtv = (unsigned short*)alloc((size_t)H * E * D * 2);
  unsigned short* wot = (unsigned short*)alloc((size_t)HD * E * 2);        // 2 MB
  unsigned short* Qb  = (unsigned short*)alloc((size_t)B * H * S * D * 2); // 8 MB
  unsigned short* Kb  = (unsigned short*)alloc((size_t)B * H * S * D * 2);
  unsigned short* Vt  = (unsigned short*)alloc((size_t)B * H * S * D * 2);
  unsigned short* ctxb= (unsigned short*)alloc((size_t)B * S * HD * 2);    // 8 MB
  unsigned int*   mb  = (unsigned int*)  alloc((size_t)S * SW * 4);        // 512 KB

  // 1) convert/transpose + mask bit-pack: 4M + 3M + 1M + 4M = 12M threads
  {
    const long total = (long)B * S * E + 3L * H * E * D + (long)HD * E + (long)S * S;
    const int blocks = (int)((total + 255) / 256);
    cvt_kernel<<<blocks, 256, 0, stream>>>(x, mask, Wq, Wk, Wv, Wo,
                                           xb, wtq, wtk, wtv, wot, mb);
  }
  // 2) Q/K/V projections: 3 * B*H*(S/16)*(D/16) = 49152 waves, 8/block
  qkv_kernel<<<49152 / 8, 256, 0, stream>>>(xb, wtq, wtk, wtv, bq, bk, bv,
                                            Qb, Kb, Vt);
  // 3) flash attention: B*H*(S/64) blocks of 4 waves = 1024 blocks
  attn_kernel<<<1024, 128, 0, stream>>>(Qb, Kb, Vt, mb, ctxb);
  // 4) output projection: (B*S/16)*(E/16) = 16384 waves, 8/block
  oproj_kernel<<<16384 / 8, 256, 0, stream>>>(ctxb, wot, bo, out);
}
